// EquivariantMPLayer_86191403696811
// MI455X (gfx1250) — matrix operand order repaired and checked by reading, not compiled, not run on hardware
//
#include <hip/hip_runtime.h>

// ---------------------------------------------------------------------------
// EquivariantMPLayer for MI455X (gfx1250, wave32, WMMA bf16 16x16x32)
//  - edge MLP as [16x257]@[257x128] per wave, W_msg staged once/block in LDS
//    via async global->LDS (ASYNCcnt) when available
//  - scatter-sum via f32 global atomics (L2-resident)
//  - node update fused: x@W_res + relu(concat(x,aggr)@W_upd + b)
// ---------------------------------------------------------------------------
typedef __bf16 bf16_t;
typedef __attribute__((ext_vector_type(16))) __bf16 v16bf;
typedef __attribute__((ext_vector_type(8)))  float  v8f;
typedef __attribute__((ext_vector_type(4)))  int    v4i;

constexpr int N_NODES = 50000;
constexpr int N_EDGES = 1600000;
constexpr int CH      = 128;   // IN_CH == HID_CH == 128
constexpr int KMSG    = 288;   // padded row stride of transposed W_msg (257 -> 288)
constexpr int KLDS    = 296;   // LDS row stride (148 dwords -> conflict-free b128)

#if defined(__has_builtin)
#if __has_builtin(__builtin_amdgcn_global_load_async_to_lds_b128) && \
    __has_builtin(__builtin_amdgcn_s_wait_asynccnt)
#define USE_ASYNC_LDS 1
#endif
#endif
#ifndef USE_ASYNC_LDS
#define USE_ASYNC_LDS 0
#endif

struct Bf16x16 { bf16_t e[16]; };
struct U4x2    { uint4 lo, hi; };

// 16 bf16 fragment from two 8-element (16B) contiguous chunks
__device__ inline v16bf load_frag16(const bf16_t* p0, const bf16_t* p1) {
  U4x2 f;
  f.lo = *reinterpret_cast<const uint4*>(p0);
  f.hi = *reinterpret_cast<const uint4*>(p1);
  return __builtin_bit_cast(v16bf, f);
}

// 16 bf16 fragment converted on the fly from two 8-float chunks
__device__ inline v16bf cvt_frag16(const float* p0, const float* p1) {
  Bf16x16 t;
#pragma unroll
  for (int i = 0; i < 8; ++i) { t.e[i] = (bf16_t)p0[i]; t.e[8 + i] = (bf16_t)p1[i]; }
  return __builtin_bit_cast(v16bf, t);
}

__device__ inline v8f wmma_bf16(v16bf a, v16bf b, v8f c) {
  // D(f32 16x16) = A(bf16 16x32) * B(bf16 32x16) + C
  return __builtin_amdgcn_wmma_f32_16x16x32_bf16(false, a, false, b, (short)0, c,
                                                 false, false);
}

#if USE_ASYNC_LDS
typedef __attribute__((address_space(1))) v4i* gv4i_p;  // global int4*
typedef __attribute__((address_space(3))) v4i* lv4i_p;  // LDS int4*
__device__ inline void async_cp16(const bf16_t* g, bf16_t* l) {
  __builtin_amdgcn_global_load_async_to_lds_b128((gv4i_p)g, (lv4i_p)l, 0, 0);
}
#endif

// --------------------------- prep kernels ----------------------------------
__global__ void k_cvt_embed(const float* __restrict__ x, bf16_t* __restrict__ y, int n) {
  int i = blockIdx.x * blockDim.x + threadIdx.x;
  if (i < n) y[i] = (bf16_t)x[i];
}

// W_msg [257][128] (row-major, [in,out]) -> Wt [128][288] bf16, zero padded
__global__ void k_prep_wmsg(const float* __restrict__ W, bf16_t* __restrict__ Wt) {
  int i = blockIdx.x * blockDim.x + threadIdx.x;
  if (i >= CH * KMSG) return;
  int n = i / KMSG, k = i % KMSG;
  Wt[i] = (k < 257) ? (bf16_t)W[k * CH + n] : (bf16_t)0.0f;
}

// W [K][128] -> Wt [128][K] bf16
__global__ void k_prep_w(const float* __restrict__ W, bf16_t* __restrict__ Wt, int K) {
  int i = blockIdx.x * blockDim.x + threadIdx.x;
  if (i >= CH * K) return;
  int n = i / K, k = i % K;
  Wt[n * K + k] = (bf16_t)W[k * CH + n];
}

__global__ void k_dist(const int* __restrict__ ei, const float* __restrict__ pos,
                       float* __restrict__ dist) {
  int e = blockIdx.x * blockDim.x + threadIdx.x;
  if (e >= N_EDGES) return;
  int r = ei[e], c = ei[N_EDGES + e];
  float dx = pos[r * 3 + 0] - pos[c * 3 + 0];
  float dy = pos[r * 3 + 1] - pos[c * 3 + 1];
  float dz = pos[r * 3 + 2] - pos[c * 3 + 2];
  dist[e] = dx * dx + dy * dy + dz * dz;
}

// --------------------------- edge MLP + scatter ----------------------------
// One wave per 16-edge tile: [16 x 257] @ [257 x 128] -> relu -> atomic scatter.
// W_msg staged once per block into LDS (74 KB of the WGP's 320 KB).
__global__ __launch_bounds__(256) void k_edge(
    const bf16_t* __restrict__ xe,   // [N_NODES][128] bf16
    const int*    __restrict__ ei,   // [2][N_EDGES]
    const float*  __restrict__ dist, // [N_EDGES]
    const bf16_t* __restrict__ Wt,   // [128][288] bf16 (transposed, padded)
    const float*  __restrict__ bmsg, // [128]
    float*        __restrict__ aggr) // [N_NODES][128] f32 (zeroed)
{
  extern __shared__ char smem_raw[];
  bf16_t* sW = (bf16_t*)smem_raw;    // [128][KLDS]

  // ---- stage W_msg into LDS: 128 rows x 36 chunks of 16B ----
  for (int c = threadIdx.x; c < CH * 36; c += 256) {
    const int n  = c / 36, ch = c % 36;
    const bf16_t* g = Wt + n * KMSG + ch * 8;
    bf16_t*       l = sW + n * KLDS + ch * 8;
#if USE_ASYNC_LDS
    async_cp16(g, l);                 // global_load_async_to_lds_b128 (ASYNCcnt)
#else
    *reinterpret_cast<uint4*>(l) = *reinterpret_cast<const uint4*>(g);
#endif
  }
#if USE_ASYNC_LDS
  __builtin_amdgcn_s_wait_asynccnt(0);
#endif
  __syncthreads();

  const int lane = threadIdx.x & 31;
  const int wave = threadIdx.x >> 5;
  const int tileBase = (blockIdx.x * 8 + wave) * 16;
  const int m    = lane & 15;        // A-row / B-column within tile
  const int half = lane >> 4;        // 0: low half-wave, 1: high half-wave
  const int e0   = tileBase + m;
  const int src  = ei[e0];
  const int tgt  = ei[N_EDGES + e0];

  v8f acc[8] = {};                   // 8 N-tiles of 16x16 f32

#pragma unroll
  for (int ks = 0; ks < 8; ++ks) {   // K = 0..255 of concat(src, tgt)
    const int node  = (ks < 4) ? src : tgt;
    const int fBase = (ks & 3) * 32 + half * 16; // A: 16 contiguous K per lane
    const bf16_t* pa = xe + (long)node * CH + fBase;
    v16bf a = load_frag16(pa, pa + 8);
    const int kOff = ks * 32 + half * 16;        // B: 16 contiguous K per lane
#pragma unroll
    for (int nt = 0; nt < 8; ++nt) {
      const bf16_t* pb = sW + (nt * 16 + m) * KLDS + kOff;   // ds_load_b128 x2
      acc[nt] = wmma_bf16(a, load_frag16(pb, pb + 8), acc[nt]);
    }
  }

  // rank-1 dist column (K=256) + bias + relu + scatter-add
  const int mBase = half * 8;
  float d[8];
  int   cols[8];
#pragma unroll
  for (int r = 0; r < 8; ++r) {
    int e   = tileBase + mBase + r;
    d[r]    = dist[e];
    cols[r] = ei[N_EDGES + e];
  }
#pragma unroll
  for (int nt = 0; nt < 8; ++nt) {
    const int   n    = nt * 16 + m;
    const float w256 = (float)sW[n * KLDS + 256];
    const float bias = bmsg[n];
#pragma unroll
    for (int r = 0; r < 8; ++r) {
      float v = fmaxf(acc[nt][r] + d[r] * w256 + bias, 0.0f);
      atomicAdd(&aggr[(long)cols[r] * CH + n], v);   // f32 atomics stay in L2
    }
  }
}

// --------------------------- node update -----------------------------------
// out = x @ W_res + relu(concat(x, aggr) @ W_upd + b_upd)
__global__ __launch_bounds__(256) void k_node(
    const bf16_t* __restrict__ xe,   // [N_NODES][128] bf16
    const float*  __restrict__ aggr, // [N_NODES][128] f32
    const bf16_t* __restrict__ WtU,  // [128][256] bf16
    const bf16_t* __restrict__ WtR,  // [128][128] bf16
    const float*  __restrict__ bupd, // [128]
    float*        __restrict__ out)  // [N_NODES][128] f32
{
  const int lane = threadIdx.x & 31;
  const int nt   = threadIdx.x >> 5;   // N-tile 0..7
  const int nodeBase = blockIdx.x * 16;
  const int m    = lane & 15;
  const int half = lane >> 4;
  const int node = nodeBase + m;

  v8f accU = {}, accR = {};

#pragma unroll
  for (int ks = 0; ks < 4; ++ks) {     // K 0..127: embed half, shared A for both GEMMs
    const int fBase = ks * 32 + half * 16;
    const bf16_t* pa = xe + (long)node * CH + fBase;
    v16bf a = load_frag16(pa, pa + 8);
    const int kOff = ks * 32 + half * 16;
    const bf16_t* pu = WtU + (nt * 16 + m) * 256 + kOff;
    accU = wmma_bf16(a, load_frag16(pu, pu + 8), accU);
    const bf16_t* pr = WtR + (nt * 16 + m) * 128 + kOff;
    accR = wmma_bf16(a, load_frag16(pr, pr + 8), accR);
  }
#pragma unroll
  for (int ks = 0; ks < 4; ++ks) {     // K 128..255: aggr half, f32 -> bf16 inline
    const int fBase = ks * 32 + half * 16;
    const float* pa = aggr + (long)node * CH + fBase;
    v16bf a = cvt_frag16(pa, pa + 8);
    const int kOff = (ks + 4) * 32 + half * 16;
    const bf16_t* pu = WtU + (nt * 16 + m) * 256 + kOff;
    accU = wmma_bf16(a, load_frag16(pu, pu + 8), accU);
  }

  const int n     = nt * 16 + m;
  const int mBase = half * 8;
  const float bias = bupd[n];
#pragma unroll
  for (int r = 0; r < 8; ++r) {
    out[(long)(nodeBase + mBase + r) * CH + n] = accR[r] + fmaxf(accU[r] + bias, 0.0f);
  }
}

// --------------------------- launcher --------------------------------------
extern "C" void kernel_launch(void* const* d_in, const int* in_sizes, int n_in,
                              void* d_out, int out_size, void* d_ws, size_t ws_size,
                              hipStream_t stream) {
  (void)in_sizes; (void)n_in; (void)out_size; (void)ws_size;
  const float* node_embed = (const float*)d_in[0];
  const float* node_pos   = (const float*)d_in[1];
  const int*   edge_index = (const int*)d_in[2];
  const float* W_res      = (const float*)d_in[3];
  const float* W_msg      = (const float*)d_in[4];
  const float* b_msg      = (const float*)d_in[5];
  const float* W_upd      = (const float*)d_in[6];
  const float* b_upd      = (const float*)d_in[7];
  float* out = (float*)d_out;

  char* ws = (char*)d_ws;
  size_t off = 0;
  auto alloc = [&](size_t bytes) -> void* {
    off = (off + 255) & ~(size_t)255;
    void* p = ws + off;
    off += bytes;
    return p;
  };
  float*  aggr  = (float*) alloc((size_t)N_NODES * CH * sizeof(float));
  bf16_t* xe    = (bf16_t*)alloc((size_t)N_NODES * CH * sizeof(bf16_t));
  float*  dist  = (float*) alloc((size_t)N_EDGES * sizeof(float));
  bf16_t* WtMsg = (bf16_t*)alloc((size_t)CH * KMSG * sizeof(bf16_t));
  bf16_t* WtUpd = (bf16_t*)alloc((size_t)CH * 256 * sizeof(bf16_t));
  bf16_t* WtRes = (bf16_t*)alloc((size_t)CH * 128 * sizeof(bf16_t));

  (void)hipMemsetAsync(aggr, 0, (size_t)N_NODES * CH * sizeof(float), stream);

  k_cvt_embed<<<(N_NODES * CH + 255) / 256, 256, 0, stream>>>(node_embed, xe, N_NODES * CH);
  k_prep_wmsg<<<(CH * KMSG + 255) / 256, 256, 0, stream>>>(W_msg, WtMsg);
  k_prep_w  <<<(CH * 256 + 255) / 256, 256, 0, stream>>>(W_upd, WtUpd, 256);
  k_prep_w  <<<(CH * 128 + 255) / 256, 256, 0, stream>>>(W_res, WtRes, 128);
  k_dist    <<<(N_EDGES + 255) / 256, 256, 0, stream>>>(edge_index, node_pos, dist);

  // 1.6M edges / (8 waves * 16 edges) = 12500 blocks; 74 KB dynamic LDS for W_msg
  k_edge<<<N_EDGES / 128, 256, (size_t)CH * KLDS * sizeof(bf16_t), stream>>>(
      xe, edge_index, dist, WtMsg, b_msg, aggr);
  // 50000 nodes / 16 = 3125 blocks
  k_node<<<N_NODES / 16, 256, 0, stream>>>(xe, aggr, WtUpd, WtRes, b_upd, out);
}
